// RNNTanh_15564961480884
// MI455X (gfx1250) — matrix operand order repaired and checked by reading, not compiled, not run on hardware
//
#include <hip/hip_runtime.h>
#include <hip/hip_bf16.h>
#include <math.h>

// Bidirectional 2-layer tanh RNN for MI455X (gfx1250, wave32, WMMA).
//   1. Convert xs (transposed to [T,B,E]) and all weights to bf16.
//   2. pre-GEMM per layer: pre[d][t*B+b][h] = X·W_ih^T + b_ih + b_hh  (WMMA bf16->f32)
//   3. Recurrence: 8 workgroups per direction, each owning a 64-column slice.
//      w_hh slice lives in VGPRs (128 VGPRs/wave, loaded once); h (32x512 bf16)
//      lives in LDS and is exchanged each step through a double-buffered global
//      buffer (async global->LDS reload) with an atomic-counter barrier.

#define RNN_B 32
#define RNN_T 512
#define RNN_E 256
#define RNN_H 512
#define RNN_M (RNN_T * RNN_B) // 16384 rows in the pre-GEMM

typedef __attribute__((ext_vector_type(16))) __bf16 v16bf;
typedef __attribute__((ext_vector_type(8)))  float  v8f;

union BFVec {
  v16bf v;
  uint4 q[2];
};

#if __has_builtin(__builtin_amdgcn_global_load_async_to_lds_b128) && \
    __has_builtin(__builtin_amdgcn_s_wait_asynccnt)
#define RNN_USE_ASYNC_LDS 1
typedef int v4i __attribute__((vector_size(16)));
typedef __attribute__((address_space(1))) v4i* g_v4i_ptr; // global (__device__)
typedef __attribute__((address_space(3))) v4i* l_v4i_ptr; // LDS (__shared__)
#else
#define RNN_USE_ASYNC_LDS 0
#endif

__device__ __forceinline__ unsigned short f2bf(float f) {
  unsigned int u = __float_as_uint(f);
  u += 0x7FFFu + ((u >> 16) & 1u); // round-to-nearest-even
  return (unsigned short)(u >> 16);
}

// ---------------------------------------------------------------------------
// xs [B][T][E] f32  ->  xbf [T][B][E] bf16
__global__ void k_xpose_bf(const float* __restrict__ xs,
                           unsigned short* __restrict__ xbf) {
  int idx = blockIdx.x * blockDim.x + threadIdx.x;
  const int total = RNN_B * RNN_T * RNN_E;
  if (idx >= total) return;
  int e = idx & (RNN_E - 1);
  int t = (idx >> 8) & (RNN_T - 1);
  int b = idx >> 17;
  xbf[((size_t)t * RNN_B + b) * RNN_E + e] = f2bf(xs[idx]);
}

// flat f32 -> bf16
__global__ void k_f2bf(const float* __restrict__ src,
                       unsigned short* __restrict__ dst, int n) {
  int i = blockIdx.x * blockDim.x + threadIdx.x;
  if (i < n) dst[i] = f2bf(src[i]);
}

__global__ void k_zero_counters(int* __restrict__ c) { c[threadIdx.x] = 0; }

// ---------------------------------------------------------------------------
// pre[d][m][n] = sum_k A[m][k] * W[d][n][k] + b[d][0][n] + b[d][1][n]
// A: [M][K] bf16 row-major, W: [2][512][K] bf16 row-major (rows = output cols)
// One 16x16 output tile per wave, 8 waves/block.
__global__ void __launch_bounds__(256)
k_pregemm(const unsigned short* __restrict__ A,
          const unsigned short* __restrict__ W,
          const float* __restrict__ bias, // [2][2][512]
          float* __restrict__ pre,        // [2][M][512]
          int K) {
  int wave = threadIdx.x >> 5;
  int lane = threadIdx.x & 31;
  int bid  = blockIdx.x;
  int d    = bid >> 12;          // / 4096
  int rem  = bid & 4095;
  int mt   = rem >> 2;           // 1024 m-tiles
  int nb   = rem & 3;
  int nt   = nb * 8 + wave;      // 32 n-tiles
  int m0 = mt * 16, n0 = nt * 16;

  int lrow = lane & 15;
  int klo  = (lane < 16) ? 0 : 8;   // A: interleaved K halves
  int kb   = (lane < 16) ? 0 : 16;  // unused; B also loads with klo, same order as A

  const unsigned short* arow = A + (size_t)(m0 + lrow) * K;
  const unsigned short* wrow = W + ((size_t)d * RNN_H + n0 + lrow) * K;

  v8f c = {};
  for (int kc = 0; kc < K; kc += 32) {
    BFVec a, b;
    a.q[0] = *(const uint4*)(arow + kc + klo);
    a.q[1] = *(const uint4*)(arow + kc + 16 + klo);
    b.q[0] = *(const uint4*)(wrow + kc + klo);
    b.q[1] = *(const uint4*)(wrow + kc + 16 + klo);
    c = __builtin_amdgcn_wmma_f32_16x16x32_bf16(false, a.v, false, b.v,
                                                (short)0, c, false, false);
  }

  int n = n0 + lrow;
  float bs = bias[d * 2 * RNN_H + n] + bias[d * 2 * RNN_H + RNN_H + n];
  float* out = pre + ((size_t)d * RNN_M + m0) * RNN_H;
#pragma unroll
  for (int rr = 0; rr < 8; ++rr) {
    int ml = rr + ((lane < 16) ? 0 : 8);
    out[(size_t)ml * RNN_H + n] = c[rr] + bs;
  }
}

// ---------------------------------------------------------------------------
// Recurrence. grid = 16 blocks (2 dirs x 8 N-slices), 256 threads (8 waves).
// Each wave: B-matrix (16 cols x 512 K of w_hh) resident in 128 VGPRs.
// h (32x512 bf16, 32KB) in LDS, exchanged via double-buffered global hbuf.
__global__ void __launch_bounds__(256, 1)
k_recurrent(const float* __restrict__ pre,           // [2][T*32][512]
            const unsigned short* __restrict__ whh,  // [2][512][512] bf16
            unsigned short* __restrict__ hbuf,       // [2][2][32][512] bf16
            unsigned short* __restrict__ out_bf,     // [T][32][1024] or null
            float* __restrict__ out_f32,             // [32][T][1024] or null
            float* __restrict__ hy,                  // this layer: [2][32][512]
            int* __restrict__ counters) {
  __shared__ __align__(16) unsigned short h_lds[RNN_B * RNN_H];   // 32 KB

  const int tid  = threadIdx.x;
  const int wave = tid >> 5;
  const int lane = tid & 31;
  const int d     = blockIdx.x >> 3;
  const int slice = blockIdx.x & 7;
  const int n0    = slice * 64;

  const int mt = wave >> 2;       // 2 M-tiles (batch 0-15 / 16-31)
  const int nt = wave & 3;        // 4 N-tiles of 16 within the 64-col slice
  const int lrow = lane & 15;
  const int klo  = (lane < 16) ? 0 : 8;
  const int kb   = (lane < 16) ? 0 : 16;
  const int mhalf = (lane < 16) ? 0 : 8;
  const int n = n0 + nt * 16 + lrow;     // global output column

  // Load this wave's B-matrix (w_hh rows [n, K]) into registers, once.
  BFVec breg[16];
  {
    const unsigned short* wrow = whh + ((size_t)d * RNN_H + n) * RNN_H;
#pragma unroll
    for (int kc16 = 0; kc16 < 16; ++kc16) {
      breg[kc16].q[0] = *(const uint4*)(wrow + kc16 * 32 + klo);
      breg[kc16].q[1] = *(const uint4*)(wrow + kc16 * 32 + 16 + klo);
    }
  }

  // h0 = 0
  {
    uint4 z = {0u, 0u, 0u, 0u};
    uint4* hq = (uint4*)h_lds;
    for (int i = tid; i < RNN_B * RNN_H / 8; i += 256) hq[i] = z;
  }
  __syncthreads();

  const unsigned short* hrow = h_lds + (size_t)(mt * 16 + lrow) * RNN_H;

  for (int s = 0; s < RNN_T; ++s) {
    const int t = (d == 0) ? s : (RNN_T - 1 - s);

    v8f c = {};
#pragma unroll
    for (int kc16 = 0; kc16 < 16; ++kc16) {
      BFVec a;
      a.q[0] = *(const uint4*)(hrow + kc16 * 32 + klo);
      a.q[1] = *(const uint4*)(hrow + kc16 * 32 + 16 + klo);
      c = __builtin_amdgcn_wmma_f32_16x16x32_bf16(false, a.v, false,
                                                  breg[kc16].v, (short)0, c,
                                                  false, false);
    }

    // Epilogue: add pre, tanh, write new h slice + layer outputs.
    const float* prow = pre + ((size_t)d * RNN_M + (size_t)t * RNN_B) * RNN_H;
    const int par = (s + 1) & 1;
    unsigned short* hb = hbuf + ((size_t)d * 2 + par) * RNN_B * RNN_H;
#pragma unroll
    for (int rr = 0; rr < 8; ++rr) {
      const int bidx = mt * 16 + rr + mhalf;
      const float v = tanhf(c[rr] + prow[(size_t)bidx * RNN_H + n]);
      hb[(size_t)bidx * RNN_H + n] = f2bf(v);
      if (out_bf)
        out_bf[((size_t)t * RNN_B + bidx) * (2 * RNN_H) + d * RNN_H + n] = f2bf(v);
      if (out_f32)
        out_f32[((size_t)bidx * RNN_T + t) * (2 * RNN_H) + d * RNN_H + n] = v;
      if (s == RNN_T - 1)
        hy[((size_t)d * RNN_B + bidx) * RNN_H + n] = v;
    }

    // Prefetch next step's pre row while we sit in the barrier.
    if (s + 1 < RNN_T) {
      const int tn = (d == 0) ? (s + 1) : (RNN_T - 2 - s);
      const float* pnext =
          pre + ((size_t)d * RNN_M + (size_t)tn * RNN_B) * RNN_H;
      __builtin_prefetch(pnext + (size_t)(mt * 16 + mhalf) * RNN_H + n, 0, 0);
    }

    // Cross-WG barrier (8 WGs of this direction); hbuf is double-buffered so
    // one barrier per step suffices.
    __threadfence();
    __syncthreads();
    if (tid == 0) {
      __hip_atomic_fetch_add(counters + d, 1, __ATOMIC_RELEASE,
                             __HIP_MEMORY_SCOPE_AGENT);
      while (__hip_atomic_load(counters + d, __ATOMIC_ACQUIRE,
                               __HIP_MEMORY_SCOPE_AGENT) < 8 * (s + 1)) {
      }
    }
    __syncthreads();

    // Reload full h into LDS from the freshly written parity buffer.
    if (s != RNN_T - 1) {
#if RNN_USE_ASYNC_LDS
      for (int i = tid; i < RNN_B * RNN_H / 8; i += 256) {
        __builtin_amdgcn_global_load_async_to_lds_b128(
            (g_v4i_ptr)(hb + (size_t)i * 8), (l_v4i_ptr)&h_lds[(size_t)i * 8],
            0, 0);
      }
      __builtin_amdgcn_s_wait_asynccnt(0);
#else
      const uint4* hsrc = (const uint4*)hb;
      uint4* hq = (uint4*)h_lds;
      for (int i = tid; i < RNN_B * RNN_H / 8; i += 256) hq[i] = hsrc[i];
#endif
    }
    __syncthreads();
  }
}

// ---------------------------------------------------------------------------
extern "C" void kernel_launch(void* const* d_in, const int* in_sizes, int n_in,
                              void* d_out, int out_size, void* d_ws,
                              size_t ws_size, hipStream_t stream) {
  const float* xs   = (const float*)d_in[0];
  const float* wih0 = (const float*)d_in[1];
  const float* whh0 = (const float*)d_in[2];
  const float* b0   = (const float*)d_in[3];
  const float* wih1 = (const float*)d_in[4];
  const float* whh1 = (const float*)d_in[5];
  const float* b1   = (const float*)d_in[6];

  float* hs = (float*)d_out;                              // [32][512][1024]
  float* hy = hs + (size_t)RNN_B * RNN_T * 2 * RNN_H;     // [4][32][512]

  char* ws = (char*)d_ws;
  size_t off = 0;
  auto alloc = [&](size_t bytes) -> void* {
    void* p = ws + off;
    off += (bytes + 255) & ~(size_t)255;
    return p;
  };
  unsigned short* xbf    = (unsigned short*)alloc((size_t)RNN_T * RNN_B * RNN_E * 2);
  unsigned short* wih0bf = (unsigned short*)alloc((size_t)2 * RNN_H * RNN_E * 2);
  unsigned short* whh0bf = (unsigned short*)alloc((size_t)2 * RNN_H * RNN_H * 2);
  unsigned short* wih1bf = (unsigned short*)alloc((size_t)2 * RNN_H * 2 * RNN_H * 2);
  unsigned short* whh1bf = (unsigned short*)alloc((size_t)2 * RNN_H * RNN_H * 2);
  unsigned short* h1bf   = (unsigned short*)alloc((size_t)RNN_T * RNN_B * 2 * RNN_H * 2);
  float*          pre    = (float*)alloc((size_t)2 * RNN_M * RNN_H * 4);
  unsigned short* hbuf   = (unsigned short*)alloc((size_t)2 * 2 * RNN_B * RNN_H * 2);
  int*            cnt    = (int*)alloc(64 * sizeof(int));

  // 1. Conversions.
  k_xpose_bf<<<(RNN_B * RNN_T * RNN_E + 255) / 256, 256, 0, stream>>>(xs, xbf);
  k_f2bf<<<(2 * RNN_H * RNN_E + 255) / 256, 256, 0, stream>>>(wih0, wih0bf, 2 * RNN_H * RNN_E);
  k_f2bf<<<(2 * RNN_H * RNN_H + 255) / 256, 256, 0, stream>>>(whh0, whh0bf, 2 * RNN_H * RNN_H);
  k_f2bf<<<(2 * RNN_H * 2 * RNN_H + 255) / 256, 256, 0, stream>>>(wih1, wih1bf, 2 * RNN_H * 2 * RNN_H);
  k_f2bf<<<(2 * RNN_H * RNN_H + 255) / 256, 256, 0, stream>>>(whh1, whh1bf, 2 * RNN_H * RNN_H);

  // 2. Layer 0: pre-GEMM + recurrence (writes h1bf and hy[0:2]).
  k_pregemm<<<8192, 256, 0, stream>>>(xbf, wih0bf, b0, pre, RNN_E);
  k_zero_counters<<<1, 64, 0, stream>>>(cnt);
  k_recurrent<<<16, 256, 0, stream>>>(pre, whh0bf, hbuf, h1bf, (float*)nullptr,
                                      hy, cnt);

  // 3. Layer 1: pre-GEMM over h1 (K=1024) + recurrence (writes hs and hy[2:4]).
  k_pregemm<<<8192, 256, 0, stream>>>(h1bf, wih1bf, b1, pre, 2 * RNN_H);
  k_zero_counters<<<1, 64, 0, stream>>>(cnt);
  k_recurrent<<<16, 256, 0, stream>>>(pre, whh1bf, hbuf,
                                      (unsigned short*)nullptr, hs,
                                      hy + (size_t)2 * RNN_B * RNN_H, cnt);
}
